// RobertaBiaffineDependencyParser_54443005444124
// MI455X (gfx1250) — matrix operand order repaired and checked by reading, not compiled, not run on hardware
//
#include <hip/hip_runtime.h>
#include <math.h>

#define B_  64
#define L_  512
#define D_  768
#define H_  256
#define R_  38
#define NEGV (-10000.0f)

typedef __attribute__((ext_vector_type(16))) __bf16 bf16x16;
typedef __attribute__((ext_vector_type(8)))  __bf16 bf16x8;
typedef __attribute__((ext_vector_type(8)))  float  f32x8;

__device__ __forceinline__ f32x8 wmma_bf16(bf16x16 a, bf16x16 b, f32x8 c) {
  // 8 args: (neg_a, A, neg_b, B, c_mod, C, reuse_a, reuse_b)
  return __builtin_amdgcn_wmma_f32_16x16x32_bf16(false, a, false, b, (short)0, c, false, false);
}

// A fragment: 16x32 bf16 tile, rows row0..+15 row-major (leading dim ld), K block [k0,k0+32).
// ISA layout: lanes 0-15 -> row=lane, K {0..7,16..23}; lanes 16-31 -> row=lane-16, K {8..15,24..31}.
__device__ __forceinline__ bf16x16 load_a_bf16(const __bf16* base, int ld, int row0, int k0) {
  int lane = threadIdx.x & 31;
  int r = lane & 15, h = lane >> 4;
  const __bf16* p = base + (size_t)(row0 + r) * ld + k0 + h * 8;
  bf16x8 lo = *(const bf16x8*)p;
  bf16x8 hi = *(const bf16x8*)(p + 16);
  bf16x16 a;
  #pragma unroll
  for (int i = 0; i < 8; ++i) { a[i] = lo[i]; a[8 + i] = hi[i]; }
  return a;
}

// Same but source rows are fp32; hardware packed-convert to bf16 on the fly.
__device__ __forceinline__ bf16x16 load_a_f32(const float* base, int ld, int row0, int k0) {
  int lane = threadIdx.x & 31;
  int r = lane & 15, h = lane >> 4;
  const float* p = base + (size_t)(row0 + r) * ld + k0 + h * 8;
  bf16x16 a;
  #pragma unroll
  for (int i = 0; i < 8; ++i) { a[i] = (__bf16)p[i]; a[8 + i] = (__bf16)p[16 + i]; }
  return a;
}

// B fragment: 32x16 tile; storage is "N-major" (each output col contiguous in K): buf[col*ld + k].
// Lanes 0-15 -> col=lane, K {0..7,16..23}; lanes 16-31 -> col=lane-16, K {8..15,24..31}.
__device__ __forceinline__ bf16x16 load_b_bf16(const __bf16* base, int ld, int col0, int k0) {
  int lane = threadIdx.x & 31;
  int n = lane & 15, h = lane >> 4;
  const __bf16* p = base + (size_t)(col0 + n) * ld + k0 + h * 8;
  bf16x8 lo = *(const bf16x8*)p;
  bf16x8 hi = *(const bf16x8*)(p + 16);
  bf16x16 b;
  #pragma unroll
  for (int i = 0; i < 8; ++i) { b[i] = lo[i]; b[8 + i] = hi[i]; }
  return b;
}

// ---------------------------------------------------------------------------
// acc layout: acc[0]=arc_nll_sum (f32), acc[1]=rel_nll_sum (f32), ((u32*)acc)[2]=n_sel
__global__ void zero_acc_kernel(float* acc) {
  if (threadIdx.x == 0) { acc[0] = 0.f; acc[1] = 0.f; ((unsigned*)acc)[2] = 0u; }
}

// Convert weights to bf16; transpose arcU/relU so B-operand columns are K-contiguous.
__global__ void prep_weights_kernel(const float* Wd, const float* Wh,
                                    const float* arcU, const float* relU,
                                    __bf16* Wdb, __bf16* Whb,
                                    __bf16* arcUT, __bf16* relUT) {
  const long NWD = (long)H_ * D_;
  const long NAU = (long)H_ * H_;
  const long NRU = (long)R_ * H_ * H_;
  const long total = 2 * NWD + NAU + NRU;
  for (long idx = blockIdx.x * (long)blockDim.x + threadIdx.x; idx < total;
       idx += (long)gridDim.x * blockDim.x) {
    if (idx < NWD) {
      Wdb[idx] = (__bf16)Wd[idx];
    } else if (idx < 2 * NWD) {
      Whb[idx - NWD] = (__bf16)Wh[idx - NWD];
    } else if (idx < 2 * NWD + NAU) {
      long t = idx - 2 * NWD;
      int i = (int)(t / H_), j = (int)(t % H_);
      arcUT[(size_t)j * H_ + i] = (__bf16)arcU[t];            // [j][i]
    } else {
      long t = idx - 2 * NWD - NAU;
      int o = (int)(t / ((long)H_ * H_));
      int rem = (int)(t % ((long)H_ * H_));
      int i = rem / H_, j = rem % H_;
      relUT[((size_t)o * H_ + j) * H_ + i] = (__bf16)relU[t];  // [o][j][i]
    }
  }
}

// Per-batch serial rank/cumsum -> gold heads, selection mask, rel targets.
__global__ void targets_kernel(const int* mask, const int* heads, const int* deprels,
                               int* ghc, int* sel, int* reltgt, float* acc) {
  int b = blockIdx.x;
  if (threadIdx.x != 0) return;
  int rank = -1, cnt = 0;
  for (int l = 0; l < L_; ++l) {
    int v = mask[b * L_ + l] > 0;
    if (v) rank++;
    int rc = rank < 0 ? 0 : (rank > L_ - 1 ? L_ - 1 : rank);
    int gh = heads[b * L_ + rc];
    int s = (v && gh >= 0 && gh < L_) ? 1 : 0;
    int g = gh < 0 ? 0 : (gh > L_ - 1 ? L_ - 1 : gh);
    ghc[b * L_ + l] = g;
    sel[b * L_ + l] = s;
    int rt = deprels[b * L_ + rc];
    reltgt[b * L_ + l] = rt < 0 ? 0 : (rt > R_ - 1 ? R_ - 1 : rt);
    cnt += s;
  }
  atomicAdd((unsigned*)acc + 2, (unsigned)cnt);
}

// dep_h / head_h = relu(enc @ W^T + b) as bf16. One wave per 16x64 output tile,
// software-pipelined: next K-step's A + 4 B fragments load behind current wmmas.
__global__ void proj_kernel(const float* enc, const __bf16* Wdb, const __bf16* Whb,
                            const float* bd, const float* bh,
                            __bf16* depb, __bf16* headb) {
  int tile = blockIdx.x * (blockDim.x >> 5) + (threadIdx.x >> 5);
  const int MT = (B_ * L_) / 16;   // 2048
  const int NG = H_ / 64;          // 4 col-groups of 64
  int which = tile / (MT * NG);
  int rem = tile % (MT * NG);
  int mt = rem / NG, ng = rem % NG;
  const __bf16* W = which ? Whb : Wdb;
  const float* bias = which ? bh : bd;
  __bf16* out = which ? headb : depb;
  int m0 = mt * 16, n0 = ng * 64;

  f32x8 c[4] = {};
  bf16x16 a = load_a_f32(enc, D_, m0, 0);
  bf16x16 b[4];
  #pragma unroll
  for (int t = 0; t < 4; ++t) b[t] = load_b_bf16(W, D_, n0 + t * 16, 0);

  for (int k0 = 0; k0 < D_ - 32; k0 += 32) {
    bf16x16 an = load_a_f32(enc, D_, m0, k0 + 32);
    bf16x16 bn[4];
    #pragma unroll
    for (int t = 0; t < 4; ++t) bn[t] = load_b_bf16(W, D_, n0 + t * 16, k0 + 32);
    #pragma unroll
    for (int t = 0; t < 4; ++t) c[t] = wmma_bf16(a, b[t], c[t]);
    a = an;
    #pragma unroll
    for (int t = 0; t < 4; ++t) b[t] = bn[t];
  }
  #pragma unroll
  for (int t = 0; t < 4; ++t) c[t] = wmma_bf16(a, b[t], c[t]);

  int lane = threadIdx.x & 31, nn = lane & 15, h = lane >> 4;
  #pragma unroll
  for (int t = 0; t < 4; ++t) {
    #pragma unroll
    for (int r = 0; r < 8; ++r) {
      int row = m0 + r + h * 8, col = n0 + t * 16 + nn;
      float v = c[t][r] + bias[col];
      v = v > 0.f ? v : 0.f;
      out[(size_t)row * H_ + col] = (__bf16)v;
    }
  }
}

// w1[t] = dep.arcW1 ; w2[t] = head.arcW2 + arcb
__global__ void linterms_kernel(const __bf16* depb, const __bf16* headb,
                                const float* arcW1, const float* arcW2, const float* arcb,
                                float* w1, float* w2) {
  int t = blockIdx.x * blockDim.x + threadIdx.x;
  if (t >= B_ * L_) return;
  float s1 = 0.f, s2 = 0.f;
  for (int i = 0; i < H_; ++i) {
    s1 += (float)depb[(size_t)t * H_ + i] * arcW1[i];
    s2 += (float)headb[(size_t)t * H_ + i] * arcW2[i];
  }
  w1[t] = s1;
  w2[t] = s2 + arcb[0];
}

// T = dep @ arcU  (bf16 out), 16x64 tile per wave, pipelined.
__global__ void arcT_kernel(const __bf16* depb, const __bf16* arcUT, __bf16* Tb) {
  int tile = blockIdx.x * (blockDim.x >> 5) + (threadIdx.x >> 5);
  const int NG = H_ / 64;          // 4
  int mt = tile / NG, ng = tile % NG;
  int m0 = mt * 16, n0 = ng * 64;

  f32x8 c[4] = {};
  bf16x16 a = load_a_bf16(depb, H_, m0, 0);
  bf16x16 b[4];
  #pragma unroll
  for (int t = 0; t < 4; ++t) b[t] = load_b_bf16(arcUT, H_, n0 + t * 16, 0);

  for (int k0 = 0; k0 < H_ - 32; k0 += 32) {
    bf16x16 an = load_a_bf16(depb, H_, m0, k0 + 32);
    bf16x16 bn[4];
    #pragma unroll
    for (int t = 0; t < 4; ++t) bn[t] = load_b_bf16(arcUT, H_, n0 + t * 16, k0 + 32);
    #pragma unroll
    for (int t = 0; t < 4; ++t) c[t] = wmma_bf16(a, b[t], c[t]);
    a = an;
    #pragma unroll
    for (int t = 0; t < 4; ++t) b[t] = bn[t];
  }
  #pragma unroll
  for (int t = 0; t < 4; ++t) c[t] = wmma_bf16(a, b[t], c[t]);

  int lane = threadIdx.x & 31, nn = lane & 15, h = lane >> 4;
  #pragma unroll
  for (int t = 0; t < 4; ++t)
    #pragma unroll
    for (int r = 0; r < 8; ++r)
      Tb[(size_t)(m0 + r + h * 8) * H_ + n0 + t * 16 + nn] = (__bf16)c[t][r];
}

// arc[b,l,m] = T[b,l,:].head[b,m,:] + w1 + w2, masked on key dim m. 16x64 per wave.
__global__ void arc_kernel(const __bf16* Tb, const __bf16* headb,
                           const float* w1, const float* w2, const int* mask, float* out) {
  int tile = blockIdx.x * (blockDim.x >> 5) + (threadIdx.x >> 5);
  const int LT = L_ / 16;   // 32 query tiles
  const int MG = L_ / 64;   // 8 key groups of 64
  int b0 = tile / (LT * MG);
  int rem = tile % (LT * MG);
  int lt = rem / MG, mg = rem % MG;
  int l0 = lt * 16, m0 = mg * 64;
  const __bf16* A = Tb + (size_t)b0 * L_ * H_;
  const __bf16* Bm = headb + (size_t)b0 * L_ * H_;  // head rows = K-contiguous cols

  f32x8 c[4] = {};
  bf16x16 a = load_a_bf16(A, H_, l0, 0);
  bf16x16 b[4];
  #pragma unroll
  for (int t = 0; t < 4; ++t) b[t] = load_b_bf16(Bm, H_, m0 + t * 16, 0);

  for (int k0 = 0; k0 < H_ - 32; k0 += 32) {
    bf16x16 an = load_a_bf16(A, H_, l0, k0 + 32);
    bf16x16 bn[4];
    #pragma unroll
    for (int t = 0; t < 4; ++t) bn[t] = load_b_bf16(Bm, H_, m0 + t * 16, k0 + 32);
    #pragma unroll
    for (int t = 0; t < 4; ++t) c[t] = wmma_bf16(a, b[t], c[t]);
    a = an;
    #pragma unroll
    for (int t = 0; t < 4; ++t) b[t] = bn[t];
  }
  #pragma unroll
  for (int t = 0; t < 4; ++t) c[t] = wmma_bf16(a, b[t], c[t]);

  int lane = threadIdx.x & 31, nn = lane & 15, h = lane >> 4;
  #pragma unroll
  for (int t = 0; t < 4; ++t) {
    #pragma unroll
    for (int r = 0; r < 8; ++r) {
      int l = l0 + r + h * 8, m = m0 + t * 16 + nn;
      float v = c[t][r] + w1[b0 * L_ + l] + w2[b0 * L_ + m];
      if (mask[b0 * L_ + m] <= 0) v = NEGV;
      out[(size_t)b0 * L_ * L_ + (size_t)l * L_ + m] = v;
    }
  }
}

// Row log-softmax NLL over arc scores.
__global__ void arc_loss_kernel(const float* scores, const int* ghc, const int* sel, float* acc) {
  __shared__ float red[256];
  int t = blockIdx.x;                          // token = b*L + l; row offset = t*L
  const float* row = scores + (size_t)t * L_;
  int tid = threadIdx.x;
  float mx = -3.0e38f;
  for (int j = tid; j < L_; j += blockDim.x) mx = fmaxf(mx, row[j]);
  red[tid] = mx; __syncthreads();
  for (int s = blockDim.x / 2; s > 0; s >>= 1) {
    if (tid < s) red[tid] = fmaxf(red[tid], red[tid + s]);
    __syncthreads();
  }
  mx = red[0]; __syncthreads();
  float sum = 0.f;
  for (int j = tid; j < L_; j += blockDim.x) sum += __expf(row[j] - mx);
  red[tid] = sum; __syncthreads();
  for (int s = blockDim.x / 2; s > 0; s >>= 1) {
    if (tid < s) red[tid] += red[tid + s];
    __syncthreads();
  }
  if (tid == 0 && sel[t]) atomicAdd(acc + 0, __logf(red[0]) + mx - row[ghc[t]]);
}

// Rel biaffine: per 16-token tile, 4 waves split the 38 relations.
// X = dep_tile @ relU[o]; rel_logit[m,o] = sum_j X[m,j]*head_vec[m,j] (+ linear terms).
__global__ void __launch_bounds__(128) rel_kernel(
    const __bf16* depb, const __bf16* headb, const __bf16* relUT,
    const float* relW1, const float* relW2, const float* relb,
    const int* ghc, const int* sel, const int* reltgt, float* acc) {
  __shared__ __bf16 hv[16 * H_];
  __shared__ float relbuf[16][R_ + 2];
  int t0 = blockIdx.x * 16;                    // global token base (tile stays in one batch)
  int tid = threadIdx.x;

  for (int idx = tid; idx < 16 * H_; idx += blockDim.x) {
    int m = idx >> 8, j = idx & (H_ - 1);
    int tok = t0 + m;
    int src = (tok / L_) * L_ + ghc[tok];      // gathered gold-head row
    hv[idx] = headb[(size_t)src * H_ + j];
  }
  __syncthreads();

  int wave = tid >> 5, lane = tid & 31, nn = lane & 15, h = lane >> 4;

  bf16x16 afr[H_ / 32];                        // preload A (dep rows t0..t0+15)
  #pragma unroll
  for (int ks = 0; ks < H_ / 32; ++ks) afr[ks] = load_a_bf16(depb, H_, t0, ks * 32);

  for (int o = wave; o < R_; o += 4) {
    const __bf16* U = relUT + (size_t)o * H_ * H_;
    if (o + 4 < R_)                            // warm caches for next relation's weights
      __builtin_prefetch(U + (size_t)4 * H_ * H_ + lane * 64, 0, 0);
    float pacc[8] = {0, 0, 0, 0, 0, 0, 0, 0};
    for (int jt = 0; jt < H_ / 16; ++jt) {
      f32x8 x = {};
      bf16x16 bcur = load_b_bf16(U, H_, jt * 16, 0);
      #pragma unroll
      for (int ks = 0; ks < H_ / 32; ++ks) {   // pipelined: load B(ks+1) behind wmma(ks)
        bf16x16 bnext = bcur;
        if (ks + 1 < H_ / 32) bnext = load_b_bf16(U, H_, jt * 16, (ks + 1) * 32);
        x = wmma_bf16(afr[ks], bcur, x);
        bcur = bnext;
      }
      #pragma unroll
      for (int r = 0; r < 8; ++r)
        pacc[r] += x[r] * (float)hv[(r + h * 8) * H_ + jt * 16 + nn];
    }
    #pragma unroll
    for (int r = 0; r < 8; ++r) {              // reduce across the 16-lane half (N dim)
      float v = pacc[r];
      v += __shfl_xor(v, 1);
      v += __shfl_xor(v, 2);
      v += __shfl_xor(v, 4);
      v += __shfl_xor(v, 8);
      if (nn == 0) relbuf[r + h * 8][o] = v;
    }
  }
  __syncthreads();

  for (int p = tid; p < 16 * R_; p += blockDim.x) {   // linear terms
    int m = p / R_, o = p % R_;
    const __bf16* dp = depb + (size_t)(t0 + m) * H_;
    float s = relb[o];
    for (int i = 0; i < H_; ++i) {
      s += (float)dp[i] * relW1[o * H_ + i];
      s += (float)hv[m * H_ + i] * relW2[o * H_ + i];
    }
    relbuf[m][o] += s;
  }
  __syncthreads();

  if (tid < 16) {
    int tok = t0 + tid;
    if (sel[tok]) {
      float mx = relbuf[tid][0];
      for (int o = 1; o < R_; ++o) mx = fmaxf(mx, relbuf[tid][o]);
      float sum = 0.f;
      for (int o = 0; o < R_; ++o) sum += __expf(relbuf[tid][o] - mx);
      atomicAdd(acc + 1, __logf(sum) + mx - relbuf[tid][reltgt[tok]]);
    }
  }
}

__global__ void finalize_kernel(const float* acc, float* out_loss) {
  if (threadIdx.x == 0 && blockIdx.x == 0) {
    unsigned n = ((const unsigned*)acc)[2];
    float denom = (float)(n > 0 ? n : 1u);
    float a = acc[0] / denom;
    float r = n > 0 ? acc[1] / denom : 0.f;
    out_loss[0] = a + r;
  }
}

// ---------------------------------------------------------------------------
extern "C" void kernel_launch(void* const* d_in, const int* in_sizes, int n_in,
                              void* d_out, int out_size, void* d_ws, size_t ws_size,
                              hipStream_t stream) {
  (void)in_sizes; (void)n_in; (void)out_size; (void)ws_size;
  const float* enc     = (const float*)d_in[0];
  const int*   amask   = (const int*)d_in[1];
  const int*   heads   = (const int*)d_in[2];
  const int*   deprels = (const int*)d_in[3];
  const float* Wd      = (const float*)d_in[4];
  const float* bd      = (const float*)d_in[5];
  const float* Wh      = (const float*)d_in[6];
  const float* bh      = (const float*)d_in[7];
  const float* arcU    = (const float*)d_in[8];
  const float* arcW1   = (const float*)d_in[9];
  const float* arcW2   = (const float*)d_in[10];
  const float* arcb    = (const float*)d_in[11];
  const float* relU    = (const float*)d_in[12];
  const float* relW1   = (const float*)d_in[13];
  const float* relW2   = (const float*)d_in[14];
  const float* relb    = (const float*)d_in[15];

  float* scores = (float*)d_out;                 // [B, L, L]
  float* loss   = scores + (size_t)B_ * L_ * L_; // final scalar

  char* ws = (char*)d_ws;
  size_t off = 0;
  auto take = [&](size_t bytes) -> void* {
    void* p = ws + off;
    off += bytes;
    off = (off + 255) & ~(size_t)255;
    return p;
  };
  __bf16* depb   = (__bf16*)take((size_t)B_ * L_ * H_ * 2);
  __bf16* headb  = (__bf16*)take((size_t)B_ * L_ * H_ * 2);
  __bf16* Tb     = (__bf16*)take((size_t)B_ * L_ * H_ * 2);
  __bf16* Wdb    = (__bf16*)take((size_t)H_ * D_ * 2);
  __bf16* Whb    = (__bf16*)take((size_t)H_ * D_ * 2);
  __bf16* arcUTb = (__bf16*)take((size_t)H_ * H_ * 2);
  __bf16* relUTb = (__bf16*)take((size_t)R_ * H_ * H_ * 2);
  float* w1     = (float*)take((size_t)B_ * L_ * 4);
  float* w2     = (float*)take((size_t)B_ * L_ * 4);
  int*   ghc    = (int*)take((size_t)B_ * L_ * 4);
  int*   sel    = (int*)take((size_t)B_ * L_ * 4);
  int*   reltgt = (int*)take((size_t)B_ * L_ * 4);
  float* acc    = (float*)take(64);

  zero_acc_kernel<<<1, 32, 0, stream>>>(acc);

  long prep_total = 2L * H_ * D_ + (long)H_ * H_ + (long)R_ * H_ * H_;
  int prep_blocks = (int)((prep_total + 255) / 256);
  prep_weights_kernel<<<prep_blocks, 256, 0, stream>>>(Wd, Wh, arcU, relU,
                                                       Wdb, Whb, arcUTb, relUTb);

  targets_kernel<<<B_, 32, 0, stream>>>(amask, heads, deprels, ghc, sel, reltgt, acc);

  int proj_tiles = 2 * ((B_ * L_) / 16) * (H_ / 64);     // 16384 wave-tiles
  proj_kernel<<<proj_tiles / 8, 256, 0, stream>>>(enc, Wdb, Whb, bd, bh, depb, headb);

  linterms_kernel<<<(B_ * L_ + 255) / 256, 256, 0, stream>>>(depb, headb, arcW1, arcW2,
                                                             arcb, w1, w2);

  int arcT_tiles = ((B_ * L_) / 16) * (H_ / 64);         // 8192
  arcT_kernel<<<arcT_tiles / 8, 256, 0, stream>>>(depb, arcUTb, Tb);

  int arc_tiles = B_ * (L_ / 16) * (L_ / 64);            // 16384
  arc_kernel<<<arc_tiles / 8, 256, 0, stream>>>(Tb, headb, w1, w2, amask, scores);

  arc_loss_kernel<<<B_ * L_, 256, 0, stream>>>(scores, ghc, sel, acc);

  rel_kernel<<<(B_ * L_) / 16, 128, 0, stream>>>(depb, headb, relUTb, relW1, relW2,
                                                 relb, ghc, sel, reltgt, acc);

  finalize_kernel<<<1, 32, 0, stream>>>(acc, loss);
}
